// GAT_79645873537468
// MI455X (gfx1250) — matrix operand order repaired and checked by reading, not compiled, not run on hardware
//
#include <hip/hip_runtime.h>

// ---------------------------------------------------------------------------
// GAT 2-layer forward for MI455X (gfx1250, wave32, WMMA)
// ---------------------------------------------------------------------------
typedef __bf16 bf16_t;
typedef __attribute__((ext_vector_type(16))) __bf16 v16bf;
typedef __attribute__((ext_vector_type(8)))  float  v8f;

#define IN_CH 128
#define HID_TOT 256   // 4 heads * 64
#define OUT_CH 64
#define NEG_SLOPE 0.2f

// ---------------- monotonic float<->uint key for atomic float max ----------
__device__ __forceinline__ unsigned fkey(float f) {
  unsigned u = __float_as_uint(f);
  return (u & 0x80000000u) ? ~u : (u | 0x80000000u);
}
__device__ __forceinline__ float unfkey(unsigned k) {
  unsigned u = (k & 0x80000000u) ? (k & 0x7FFFFFFFu) : ~k;
  return __uint_as_float(u);
}

// ---------------- fills ----------------------------------------------------
__global__ void gat_fill_f32(float* __restrict__ p, size_t n, float v) {
  size_t i = (size_t)blockIdx.x * blockDim.x + threadIdx.x;
  if (i < n) p[i] = v;
}
__global__ void gat_fill_u32(unsigned* __restrict__ p, size_t n, unsigned v) {
  size_t i = (size_t)blockIdx.x * blockDim.x + threadIdx.x;
  if (i < n) p[i] = v;
}
// d_out[i] = b[i % C]  (C power of two) -- bias pre-load so scatter adds on top
__global__ void gat_bias_init(float* __restrict__ out, const float* __restrict__ b,
                              size_t total, int cmask) {
  size_t i = (size_t)blockIdx.x * blockDim.x + threadIdx.x;
  if (i < total) out[i] = b[i & cmask];
}

// ---------------- pack W[K,NC] f32 -> WMMA B-fragment-native bf16 ----------
// Bpk[(kblk*NC + col)*32 + k] = bf16( W[(kblk*32 + k)*NC + col] )
// => lane (col, half) reads 16 contiguous bf16 (32B) at ((kblk*NC+col)*32 + half*16)
__global__ void gat_pack_B(const float* __restrict__ W, bf16_t* __restrict__ out,
                           int K, int NC) {
  int i = blockIdx.x * blockDim.x + threadIdx.x;
  if (i >= K * NC) return;
  int kblk = i / (NC * 32);
  int rem  = i - kblk * (NC * 32);
  int col  = rem >> 5;
  int k    = rem & 31;
  out[i] = (bf16_t)W[(size_t)(kblk * 32 + k) * NC + col];
}

// ---------------- WMMA GEMM: H[n,NC] = X[n,K] @ Wpk ------------------------
// one wave computes a 16(M) x 16*NTW(N) strip; K consumed in 32-wide bf16 steps
template <int K, int NC, int NTW>
__global__ __launch_bounds__(256) void gat_gemm_wmma(
    const float* __restrict__ X, const bf16_t* __restrict__ Bpk,
    float* __restrict__ H, int mtiles) {
  const int groups = NC / (16 * NTW);
  int wave = blockIdx.x * (blockDim.x >> 5) + (threadIdx.x >> 5);
  if (wave >= mtiles * groups) return;
  int mt = wave / groups;
  int g  = wave % groups;
  int lane = threadIdx.x & 31;
  int half = lane >> 4;        // which 16-lane half
  int r    = lane & 15;
  const float* xrow = X + (size_t)(mt * 16 + r) * K;

  v8f acc[NTW];
#pragma unroll
  for (int j = 0; j < NTW; ++j) acc[j] = (v8f){0.f,0.f,0.f,0.f,0.f,0.f,0.f,0.f};

  for (int k0 = 0; k0 < K; k0 += 32) {
    // A tile 16x32 bf16: lane = row. Fragment = one contiguous 16-float run:
    // K = k0 + half*16 + [0..15]
    const float* ap = xrow + k0 + half * 16;
    float4 f0 = *(const float4*)(ap);
    float4 f1 = *(const float4*)(ap + 4);
    float4 f2 = *(const float4*)(ap + 8);
    float4 f3 = *(const float4*)(ap + 12);
    v16bf a;
    a[0]=(bf16_t)f0.x;  a[1]=(bf16_t)f0.y;  a[2]=(bf16_t)f0.z;  a[3]=(bf16_t)f0.w;
    a[4]=(bf16_t)f1.x;  a[5]=(bf16_t)f1.y;  a[6]=(bf16_t)f1.z;  a[7]=(bf16_t)f1.w;
    a[8]=(bf16_t)f2.x;  a[9]=(bf16_t)f2.y;  a[10]=(bf16_t)f2.z; a[11]=(bf16_t)f2.w;
    a[12]=(bf16_t)f3.x; a[13]=(bf16_t)f3.y; a[14]=(bf16_t)f3.z; a[15]=(bf16_t)f3.w;
#pragma unroll
    for (int j = 0; j < NTW; ++j) {
      int col = g * 16 * NTW + j * 16 + r;
      // B fragment: 16 contiguous bf16 (two global_load_b128)
      v16bf b = *(const v16bf*)(Bpk + ((size_t)(k0 >> 5) * NC + col) * 32 + half * 16);
      acc[j] = __builtin_amdgcn_wmma_f32_16x16x32_bf16(
          false, a, false, b, (short)0, acc[j], false, false);
    }
  }
  // C/D f32 16x16: lane = col, VGPR v -> row M = v + 8*half
#pragma unroll
  for (int j = 0; j < NTW; ++j) {
    int col = g * 16 * NTW + j * 16 + r;
#pragma unroll
    for (int v = 0; v < 8; ++v)
      H[(size_t)(mt * 16 + half * 8 + v) * NC + col] = acc[j][v];
  }
}

// ---------------- per-node attention logits --------------------------------
// one wave per node: als[n,h] = sum_c h[n,h,c]*a_src[h,c]; same for ald.
template <int HEADS, int C>
__global__ __launch_bounds__(256) void gat_att_logits(
    const float* __restrict__ Hf, const float* __restrict__ a_src,
    const float* __restrict__ a_dst, float* __restrict__ als,
    float* __restrict__ ald, int n) {
  const int CPL = (HEADS * C) / 32;  // channels per lane
  const int GRP = C / CPL;           // lanes per head
  int node = blockIdx.x * (blockDim.x >> 5) + (threadIdx.x >> 5);
  if (node >= n) return;
  int lane = threadIdx.x & 31;
  const float* hp = Hf + (size_t)node * HEADS * C + lane * CPL;
  const float* as = a_src + lane * CPL;
  const float* ad = a_dst + lane * CPL;
  float ss = 0.f, sd = 0.f;
#pragma unroll
  for (int i = 0; i < CPL; ++i) {
    float v = hp[i];
    ss += v * as[i];
    sd += v * ad[i];
  }
#pragma unroll
  for (int off = GRP >> 1; off > 0; off >>= 1) {
    ss += __shfl_down(ss, off, 32);
    sd += __shfl_down(sd, off, 32);
  }
  if ((lane & (GRP - 1)) == 0) {
    int h = lane / GRP;
    als[(size_t)node * HEADS + h] = ss;
    ald[(size_t)node * HEADS + h] = sd;
  }
}

// ---------------- edge pass 1: leaky-relu score + segment max --------------
template <int HEADS>
__global__ void gat_edge_score(const long long* __restrict__ ei, int E, int E2,
                               const float* __restrict__ als,
                               const float* __restrict__ ald,
                               float* __restrict__ ebuf,
                               unsigned* __restrict__ mkeys) {
  int e = blockIdx.x * blockDim.x + threadIdx.x;
  if (e >= E2) return;
  int s, d;
  if (e < E) { s = (int)ei[e]; d = (int)ei[(size_t)E + e]; }
  else       { s = d = e - E; }                     // self loop
#pragma unroll
  for (int h = 0; h < HEADS; ++h) {
    float v = als[(size_t)s * HEADS + h] + ald[(size_t)d * HEADS + h];
    v = v > 0.f ? v : NEG_SLOPE * v;
    ebuf[(size_t)e * HEADS + h] = v;
    atomicMax(&mkeys[(size_t)d * HEADS + h], fkey(v));
  }
}

// ---------------- edge pass 2: exp(e - m) + segment sum --------------------
template <int HEADS>
__global__ void gat_edge_exp(const long long* __restrict__ ei, int E, int E2,
                             const unsigned* __restrict__ mkeys,
                             float* __restrict__ ebuf, float* __restrict__ den) {
  int e = blockIdx.x * blockDim.x + threadIdx.x;
  if (e >= E2) return;
  int d = (e < E) ? (int)ei[(size_t)E + e] : (e - E);
#pragma unroll
  for (int h = 0; h < HEADS; ++h) {
    float m  = unfkey(mkeys[(size_t)d * HEADS + h]);
    float ex = __expf(ebuf[(size_t)e * HEADS + h] - m);
    ebuf[(size_t)e * HEADS + h] = ex;
    atomicAdd(&den[(size_t)d * HEADS + h], ex);
  }
}

// ---------------- edge pass 3: out[dst] += alpha * h[src] ------------------
// CH = total channels (heads*64); each thread owns one float4 (head = c/64)
template <int CH>
__global__ void gat_edge_aggregate(const long long* __restrict__ ei, int E, int E2,
                                   const float* __restrict__ ebuf,
                                   const float* __restrict__ den,
                                   const float* __restrict__ Hsrc,
                                   float* __restrict__ out) {
  const int TPE = CH / 4;
  const int HEADS = CH >> 6;
  size_t idx = (size_t)blockIdx.x * blockDim.x + threadIdx.x;
  int e = (int)(idx / TPE);
  int t = (int)(idx % TPE);
  if (e >= E2) return;
  int s, d;
  if (e < E) { s = (int)ei[e]; d = (int)ei[(size_t)E + e]; }
  else       { s = d = e - E; }
  int c4 = t * 4;
  int h  = c4 >> 6;
  float alpha = ebuf[(size_t)e * HEADS + h] / den[(size_t)d * HEADS + h];
  float4 v = *(const float4*)(Hsrc + (size_t)s * CH + c4);
  float* o = out + (size_t)d * CH + c4;
  atomicAdd(o + 0, v.x * alpha);
  atomicAdd(o + 1, v.y * alpha);
  atomicAdd(o + 2, v.z * alpha);
  atomicAdd(o + 3, v.w * alpha);
}

// ---------------- elementwise: a = elu(a + b1[c]) --------------------------
__global__ void gat_elu_bias(float* __restrict__ a, const float* __restrict__ b,
                             size_t total, int cmask) {
  size_t i = (size_t)blockIdx.x * blockDim.x + threadIdx.x;
  if (i >= total) return;
  float v = a[i] + b[i & cmask];
  a[i] = v > 0.f ? v : (__expf(v) - 1.f);
}

// ---------------------------------------------------------------------------
extern "C" void kernel_launch(void* const* d_in, const int* in_sizes, int n_in,
                              void* d_out, int out_size, void* d_ws, size_t ws_size,
                              hipStream_t stream) {
  const float*     x      = (const float*)d_in[0];
  const long long* ei     = (const long long*)d_in[1];   // int64 [2,E]
  const float*     W1     = (const float*)d_in[2];
  const float*     a_src1 = (const float*)d_in[3];
  const float*     a_dst1 = (const float*)d_in[4];
  const float*     b1     = (const float*)d_in[5];
  const float*     W2     = (const float*)d_in[6];
  const float*     a_src2 = (const float*)d_in[7];
  const float*     a_dst2 = (const float*)d_in[8];
  const float*     b2     = (const float*)d_in[9];
  float*           out    = (float*)d_out;

  const int N  = in_sizes[0] / IN_CH;   // 50000
  const int E  = in_sizes[1] / 2;       // 800000
  const int E2 = E + N;                 // + self loops
  const int MT = N / 16;                // 16 | 50000

  // -------- workspace carve-up (256B aligned) --------
  size_t off = 0;
  auto carve = [&](size_t bytes) -> char* {
    char* p = (char*)d_ws + off;
    off = (off + bytes + 255) & ~(size_t)255;
    return p;
  };
  float*    h1    = (float*)carve((size_t)N * HID_TOT * 4);
  float*    agg1  = (float*)carve((size_t)N * HID_TOT * 4);  // layer1 output / elu'd h
  float*    h2    = (float*)carve((size_t)N * OUT_CH * 4);
  float*    als1  = (float*)carve((size_t)N * 4 * 4);
  float*    ald1  = (float*)carve((size_t)N * 4 * 4);
  unsigned* m1    = (unsigned*)carve((size_t)N * 4 * 4);
  float*    den1  = (float*)carve((size_t)N * 4 * 4);
  float*    als2  = (float*)carve((size_t)N * 4);
  float*    ald2  = (float*)carve((size_t)N * 4);
  unsigned* m2    = (unsigned*)carve((size_t)N * 4);
  float*    den2  = (float*)carve((size_t)N * 4);
  float*    ebuf1 = (float*)carve((size_t)E2 * 4 * 4);
  float*    ebuf2 = (float*)carve((size_t)E2 * 4);
  bf16_t*   W1bf  = (bf16_t*)carve((size_t)IN_CH * HID_TOT * 2);
  bf16_t*   W2bf  = (bf16_t*)carve((size_t)HID_TOT * OUT_CH * 2);

  const int T = 256;
  auto blk  = [&](size_t n) { return (int)((n + T - 1) / T); };

  // -------- weight packing + buffer init (every call: ws is not persistent)
  gat_pack_B<<<blk(IN_CH * HID_TOT), T, 0, stream>>>(W1, W1bf, IN_CH, HID_TOT);
  gat_pack_B<<<blk(HID_TOT * OUT_CH), T, 0, stream>>>(W2, W2bf, HID_TOT, OUT_CH);
  gat_fill_f32<<<blk((size_t)N * HID_TOT), T, 0, stream>>>(agg1, (size_t)N * HID_TOT, 0.f);
  gat_fill_u32<<<blk((size_t)N * 4), T, 0, stream>>>(m1, (size_t)N * 4, 0u);
  gat_fill_f32<<<blk((size_t)N * 4), T, 0, stream>>>(den1, (size_t)N * 4, 0.f);
  gat_fill_u32<<<blk(N), T, 0, stream>>>(m2, (size_t)N, 0u);
  gat_fill_f32<<<blk(N), T, 0, stream>>>(den2, (size_t)N, 0.f);
  gat_bias_init<<<blk((size_t)N * OUT_CH), T, 0, stream>>>(out, b2, (size_t)N * OUT_CH, OUT_CH - 1);

  // ================= layer 1 (4 heads x 64) =================
  {
    int waves = MT * (HID_TOT / 64);           // NTW=4 -> 4 ntile-groups
    gat_gemm_wmma<IN_CH, HID_TOT, 4><<<(waves + 7) / 8, T, 0, stream>>>(x, W1bf, h1, MT);
    gat_att_logits<4, 64><<<(N + 7) / 8, T, 0, stream>>>(h1, a_src1, a_dst1, als1, ald1, N);
    gat_edge_score<4><<<blk(E2), T, 0, stream>>>(ei, E, E2, als1, ald1, ebuf1, m1);
    gat_edge_exp<4><<<blk(E2), T, 0, stream>>>(ei, E, E2, m1, ebuf1, den1);
    gat_edge_aggregate<HID_TOT><<<blk((size_t)E2 * (HID_TOT / 4)), T, 0, stream>>>(
        ei, E, E2, ebuf1, den1, h1, agg1);
    gat_elu_bias<<<blk((size_t)N * HID_TOT), T, 0, stream>>>(agg1, b1, (size_t)N * HID_TOT, HID_TOT - 1);
  }

  // ================= layer 2 (1 head x 64) =================
  {
    int waves = MT * (OUT_CH / 64);            // one 16x64 strip per wave
    gat_gemm_wmma<HID_TOT, OUT_CH, 4><<<(waves + 7) / 8, T, 0, stream>>>(agg1, W2bf, h2, MT);
    gat_att_logits<1, 64><<<(N + 7) / 8, T, 0, stream>>>(h2, a_src2, a_dst2, als2, ald2, N);
    gat_edge_score<1><<<blk(E2), T, 0, stream>>>(ei, E, E2, als2, ald2, ebuf2, m2);
    gat_edge_exp<1><<<blk(E2), T, 0, stream>>>(ei, E, E2, m2, ebuf2, den2);
    gat_edge_aggregate<OUT_CH><<<blk((size_t)E2 * (OUT_CH / 4)), T, 0, stream>>>(
        ei, E, E2, ebuf2, den2, h2, out);      // lands on b2-initialized d_out
  }
  (void)n_in; (void)out_size; (void)ws_size;
}
